// TrivialTreeTagger_68418829025428
// MI455X (gfx1250) — matrix operand order repaired and checked by reading, not compiled, not run on hardware
//
#include <hip/hip_runtime.h>

// ---------------------------------------------------------------------------
// TrivialTreeTagger on MI455X (gfx1250, wave32, WMMA).
//
// Strategy: collapse leaf level to a 26-entry table, fold the constant first
// LSTM step into bias2/c1, fold emb-side of W_comb into a 26x64 table, then
// run 18 level kernels bottom-up. Each wave computes a 16-node tile with
// V_WMMA_F32_16X16X32_F16 (M=16 nodes, K=64, N tiled by 16 over 256 gates).
// Inter-level activations stored f16. Key codegen controls:
//  - hardware-rcp sigmoid/tanh (no IEEE division sequences)
//  - a laundered per-iteration zero OFFSET (not a laundered pointer!) defeats
//    LICM of the 640-VGPR weight-fragment set while preserving global
//    address-space inference (global_load_b128, not flat_load)
//  - sched_barrier(0) between gate groups bounds in-flight B-fragments
//  -> fits 256 VGPRs (4 waves/SIMD), no spills, loads issued near use.
// ---------------------------------------------------------------------------

typedef _Float16 f16;
typedef __attribute__((ext_vector_type(16))) _Float16 v16h;
typedef __attribute__((ext_vector_type(8)))  float    v8f;

#define WAVES 4
#define SSTR  72   // f16 staging row stride (halves): 144B, 16B aligned, bank-clean
#define TSTR  68   // f32 tag row stride (floats): 272B, 16B aligned, bank-clean

struct Frag { union { v16h h; uint4 q[2]; }; };

// Fast nonlinearities: v_exp_f32 + v_rcp_f32, no IEEE-division correction.
__device__ __forceinline__ float rcpf(float x)  { return __builtin_amdgcn_rcpf(x); }
__device__ __forceinline__ float sigf(float x)  { return rcpf(1.0f + __expf(-x)); }
__device__ __forceinline__ float tanh_(float x) {
  return __builtin_fmaf(2.0f, rcpf(1.0f + __expf(-2.0f * x)), -1.0f);
}

// Hard scheduling fence: nothing moves across (bounds register pressure).
__device__ __forceinline__ void sched_fence() { __builtin_amdgcn_sched_barrier(0); }

// Opaque zero offset, re-created every loop iteration: addresses formed as
// base + zoff are loop-variant to the optimizer (no LICM -> no spilled
// hoisted weight set), but the pointer is still based on the kernel arg, so
// address-space inference keeps the loads on the GLOBAL path.
__device__ __forceinline__ size_t launder_zero() {
  unsigned long long z = 0;
  asm volatile("" : "+s"(z));
  return (size_t)z;
}

__device__ __forceinline__ v8f splat8(float x) {
  v8f c;
#pragma unroll
  for (int e = 0; e < 8; ++e) c[e] = x;
  return c;
}

__device__ __forceinline__ v8f wmma16(const Frag& a, const Frag& b, v8f c) {
  // D = A(16x32 f16) * B(32x16 f16) + C(16x16 f32)
  return __builtin_amdgcn_wmma_f32_16x16x32_f16(false, a.h, false, b.h, (short)0, c, false, false);
}

// B fragment: prepacked per (ntile, kstep, lane) as 16 contiguous halves (32B).
__device__ __forceinline__ Frag load_bfrag(const f16* base, int nt, int kk, int lane) {
  const uint4* p = reinterpret_cast<const uint4*>(base + ((size_t)((nt * 2 + kk) * 32 + lane)) * 16);
  Frag f; f.q[0] = p[0]; f.q[1] = p[1]; return f;
}

// A fragments (both K-steps) for one row of 64 f16. Per ISA 16-bit A layout:
// lanes 0-15 hold K={0..7,16..23}(+32*kk), lanes 16-31 hold K={8..15,24..31}(+32*kk)
// => two contiguous 16B chunks per lane per kstep.
__device__ __forceinline__ void load_afrag(const f16* row, Frag a[2], int hi) {
  const uint4* p = reinterpret_cast<const uint4*>(row);
  a[0].q[0] = p[0 + hi]; a[0].q[1] = p[2 + hi];
  a[1].q[0] = p[4 + hi]; a[1].q[1] = p[6 + hi];
}

// Intra-wave LDS ordering (no cross-wave deps => no s_barrier needed).
__device__ __forceinline__ void wave_lds_fence() {
  __builtin_amdgcn_wave_barrier();
  __threadfence_block();
  __builtin_amdgcn_wave_barrier();
}

// ---------------------------------------------------------------------------
// Prep 1: pack W_ih^T, W_hh^T, W_comb[:, :64]^T, W_tag^T into f16 WMMA
// B-fragment order. B element (k,n) layout: lane L holds col n=L&15;
// half h holds K = kk*32 + (h<8?0:16) + (L<16?0:8) + (h&7).
// ---------------------------------------------------------------------------
__global__ void prep_pack(const float* __restrict__ Wih, const float* __restrict__ Whh,
                          const float* __restrict__ Wcomb, const float* __restrict__ Wtag,
                          f16* __restrict__ fih, f16* __restrict__ fhh,
                          f16* __restrict__ fch, f16* __restrict__ ftg) {
  int idx = blockIdx.x * 256 + threadIdx.x;
  if (idx >= 40960) return;
  const float* src; f16* dst; int w; int which;
  if      (idx < 16384)          { which = 0; w = idx;          src = Wih;   dst = fih; }
  else if (idx < 32768)          { which = 1; w = idx - 16384;  src = Whh;   dst = fhh; }
  else if (idx < 36864)          { which = 2; w = idx - 32768;  src = Wcomb; dst = fch; }
  else                           { which = 3; w = idx - 36864;  src = Wtag;  dst = ftg; }
  int frag = w >> 9;             // (nt, kk)
  int lane = (w >> 4) & 31;
  int h    = w & 15;
  int nt = frag >> 1, kk = frag & 1;
  int gn = nt * 16 + (lane & 15);                               // output column
  int k  = kk * 32 + ((h < 8) ? 0 : 16) + ((lane < 16) ? 0 : 8) + (h & 7);  // K index
  float v;
  if      (which == 0) v = src[gn * 64 + k];
  else if (which == 1) v = src[gn * 64 + k];
  else if (which == 2) v = src[gn * 128 + k];   // W_comb[:, :64] (h-part)
  else                 v = src[gn * 64 + k];
  dst[w] = (f16)v;
}

// ---------------------------------------------------------------------------
// Prep 2 (1 block, 256 threads): constants of the collapsed step-1 + tables.
// ---------------------------------------------------------------------------
__global__ void prep_small(const float* __restrict__ emb,   const float* __restrict__ Wih,
                           const float* __restrict__ Whh,   const float* __restrict__ bih,
                           const float* __restrict__ bhh,   const float* __restrict__ Wcomb,
                           const float* __restrict__ bcomb, const float* __restrict__ Wtag,
                           const float* __restrict__ btag,  const float* __restrict__ linit,
                           float* __restrict__ bias1, float* __restrict__ bias2,
                           float* __restrict__ c1g,  float* __restrict__ comb_emb,
                           f16* __restrict__ leaf16) {
  __shared__ float sg[256], sh1[64], swh1[64], scomb[1664], stag[1664], slse[26];
  int tid = threadIdx.x;

  float b1 = bih[tid] + bhh[tid];
  bias1[tid] = b1;
  float g = b1;
  for (int k = 0; k < 64; ++k) g += Wih[tid * 64 + k] * linit[k];
  sg[tid] = g;
  __syncthreads();

  if (tid < 64) {  // c=0, h=0 => f-gate irrelevant
    float iv = sigf(sg[tid]);
    float gv = tanh_(sg[128 + tid]);
    float ov = sigf(sg[192 + tid]);
    float c1 = iv * gv;
    float h1 = ov * tanh_(c1);
    sh1[tid] = h1; c1g[tid] = c1;
  }
  __syncthreads();

  { float b2 = b1;
    for (int k = 0; k < 64; ++k) b2 += Whh[tid * 64 + k] * sh1[k];
    bias2[tid] = b2; }

  for (int e = tid; e < 1664; e += 256) {  // comb_emb (26x64)
    int v = e >> 6, j = e & 63;
    float s = bcomb[j];
    for (int k = 0; k < 64; ++k) s += Wcomb[j * 128 + 64 + k] * emb[v * 64 + k];
    scomb[e] = s; comb_emb[e] = s;
  }
  if (tid < 64) {  // W_comb[:, :64] @ h1
    float s = 0.f;
    for (int k = 0; k < 64; ++k) s += Wcomb[tid * 128 + k] * sh1[k];
    swh1[tid] = s;
  }
  __syncthreads();

  for (int e = tid; e < 1664; e += 256) {  // leaf tag logits
    int v = e >> 6, j = e & 63;
    float s = btag[j];
    for (int k = 0; k < 64; ++k) s += Wtag[j * 64 + k] * (scomb[v * 64 + k] + swh1[k]);
    stag[e] = s;
  }
  __syncthreads();
  if (tid < 26) {
    float mx = -3.0e38f;
    for (int j = 0; j < 64; ++j) mx = fmaxf(mx, stag[tid * 64 + j]);
    float s = 0.f;
    for (int j = 0; j < 64; ++j) s += __expf(stag[tid * 64 + j] - mx);
    slse[tid] = mx + __logf(s);
  }
  __syncthreads();
  for (int e = tid; e < 1664; e += 256) leaf16[e] = (f16)(stag[e] - slse[e >> 6]);
}

// ---------------------------------------------------------------------------
// One tree level. Each wave = one 16-node tile, 112 WMMAs per tile.
// ---------------------------------------------------------------------------
__global__ __launch_bounds__(128, 4) void tree_level_kernel(
    f16* __restrict__ out16, const f16* __restrict__ leaf16,
    const int* __restrict__ ids,
    const f16* __restrict__ fih0, const f16* __restrict__ fhh0,
    const f16* __restrict__ fch0, const f16* __restrict__ ftg0,
    const float* __restrict__ bias1_0, const float* __restrict__ bias2_0,
    const float* __restrict__ c1v0,    const float* __restrict__ comb_emb0,
    const float* __restrict__ btag0,   float* __restrict__ rootOut,
    int start, int Bnodes, int ntiles, int leafChildren, int isRoot) {
  __shared__ __align__(16) f16   s_stage[WAVES][16 * SSTR];
  __shared__ __align__(16) float s_tag  [WAVES][16 * TSTR];
  __shared__ float s_lse[WAVES][16];
  __shared__ int   s_id [WAVES][16];

  const int wv    = threadIdx.x >> 5;
  const int lane  = threadIdx.x & 31;
  const int n     = lane & 15;        // C-layout column / A-layout row
  const int hi    = lane >> 4;        // which K/row half this lane owns
  const int rbase = hi ? 8 : 0;       // C-layout row base
  f16*   st = s_stage[wv];
  float* tg = s_tag[wv];
  const int last = start + Bnodes - 1;

  for (int tile = blockIdx.x * WAVES + wv; tile < ntiles; tile += gridDim.x * WAVES) {
    // Per-iteration opaque zero offset: keeps every weight/bias load inside
    // the loop (no LICM -> no spills) while staying on the global-load path.
    const size_t z = launder_zero();
    const f16*   fih      = fih0      + z;
    const f16*   fhh      = fhh0      + z;
    const f16*   fch      = fch0      + z;
    const f16*   ftg      = ftg0      + z;
    const float* bias1    = bias1_0   + z;
    const float* bias2    = bias2_0   + z;
    const float* c1v      = c1v0      + z;
    const float* comb_emb = comb_emb0 + z;
    const float* btag     = btag0     + z;

    const int node0 = start + tile * 16;

    if (lane < 16) {
      int nd = node0 + lane; if (nd > last) nd = last;
      s_id[wv][lane] = ids[nd];
    }
    wave_lds_fence();

    // ---- A fragments for left / right child outputs (row m == n) ----
    int ndm = node0 + n; if (ndm > last) ndm = last;
    const f16* lrow; const f16* rrow;
    if (leafChildren) {
      lrow = leaf16 + (size_t)ids[2 * ndm + 1] * 64;
      rrow = leaf16 + (size_t)ids[2 * ndm + 2] * 64;
    } else {
      lrow = out16 + (size_t)(2 * ndm + 1) * 64;
      rrow = out16 + (size_t)(2 * ndm + 2) * 64;
    }
    Frag aL[2], aR[2];
    load_afrag(lrow, aL, hi);
    load_afrag(rrow, aR, hi);
    sched_fence();

    // ---- LSTM step 2 (x = left_out; h=h1 folded into bias2, c=c1) ----
    v8f c2f[4];
#pragma unroll
    for (int t = 0; t < 4; ++t) {
      v8f gi = splat8(bias2[(t     ) * 16 + n]);
      v8f gf = splat8(bias2[(t +  4) * 16 + n]);
      v8f gg = splat8(bias2[(t +  8) * 16 + n]);
      v8f go = splat8(bias2[(t + 12) * 16 + n]);
#pragma unroll
      for (int kk = 0; kk < 2; ++kk) {
        Frag b;
        b = load_bfrag(fih, t,      kk, lane); gi = wmma16(aL[kk], b, gi);
        b = load_bfrag(fih, t +  4, kk, lane); gf = wmma16(aL[kk], b, gf);
        b = load_bfrag(fih, t +  8, kk, lane); gg = wmma16(aL[kk], b, gg);
        b = load_bfrag(fih, t + 12, kk, lane); go = wmma16(aL[kk], b, go);
      }
      float c1j = c1v[t * 16 + n];
#pragma unroll
      for (int e = 0; e < 8; ++e) {
        float iv = sigf(gi[e]), fv = sigf(gf[e]), gv = tanh_(gg[e]), ov = sigf(go[e]);
        float c2 = fv * c1j + iv * gv;
        c2f[t][e] = c2;
        st[(e + rbase) * SSTR + t * 16 + n] = (f16)(ov * tanh_(c2));  // h2 -> LDS
      }
      sched_fence();
    }
    wave_lds_fence();
    Frag aH[2];
    load_afrag(st + n * SSTR, aH, hi);
    sched_fence();

    // ---- LSTM step 3 (x = right_out, h = h2) ----
#pragma unroll
    for (int t = 0; t < 4; ++t) {
      v8f gi = splat8(bias1[(t     ) * 16 + n]);
      v8f gf = splat8(bias1[(t +  4) * 16 + n]);
      v8f gg = splat8(bias1[(t +  8) * 16 + n]);
      v8f go = splat8(bias1[(t + 12) * 16 + n]);
#pragma unroll
      for (int kk = 0; kk < 2; ++kk) {
        Frag b;
        b = load_bfrag(fih, t,      kk, lane); gi = wmma16(aR[kk], b, gi);
        b = load_bfrag(fhh, t,      kk, lane); gi = wmma16(aH[kk], b, gi);
        b = load_bfrag(fih, t +  4, kk, lane); gf = wmma16(aR[kk], b, gf);
        b = load_bfrag(fhh, t +  4, kk, lane); gf = wmma16(aH[kk], b, gf);
        b = load_bfrag(fih, t +  8, kk, lane); gg = wmma16(aR[kk], b, gg);
        b = load_bfrag(fhh, t +  8, kk, lane); gg = wmma16(aH[kk], b, gg);
        b = load_bfrag(fih, t + 12, kk, lane); go = wmma16(aR[kk], b, go);
        b = load_bfrag(fhh, t + 12, kk, lane); go = wmma16(aH[kk], b, go);
      }
#pragma unroll
      for (int e = 0; e < 8; ++e) {
        float iv = sigf(gi[e]), fv = sigf(gf[e]), gv = tanh_(gg[e]), ov = sigf(go[e]);
        float c3 = fv * c2f[t][e] + iv * gv;
        st[(e + rbase) * SSTR + t * 16 + n] = (f16)(ov * tanh_(c3));  // h3 -> LDS
      }
      sched_fence();
    }
    wave_lds_fence();
    Frag aH3[2];
    load_afrag(st + n * SSTR, aH3, hi);
    sched_fence();

    // ---- combined = h3 @ Wch^T + comb_emb[id]  (result f16 -> LDS) ----
#pragma unroll
    for (int t = 0; t < 4; ++t) {
      v8f cc;
#pragma unroll
      for (int e = 0; e < 8; ++e)
        cc[e] = comb_emb[(size_t)s_id[wv][e + rbase] * 64 + t * 16 + n];
#pragma unroll
      for (int kk = 0; kk < 2; ++kk) {
        Frag b = load_bfrag(fch, t, kk, lane);
        cc = wmma16(aH3[kk], b, cc);
      }
#pragma unroll
      for (int e = 0; e < 8; ++e)
        st[(e + rbase) * SSTR + t * 16 + n] = (f16)cc[e];
      sched_fence();
    }
    wave_lds_fence();
    Frag aC[2];
    load_afrag(st + n * SSTR, aC, hi);
    sched_fence();

    // ---- tag = combined @ Wtag^T + b_tag  (f32 -> LDS) ----
#pragma unroll
    for (int t = 0; t < 4; ++t) {
      v8f tt = splat8(btag[t * 16 + n]);
#pragma unroll
      for (int kk = 0; kk < 2; ++kk) {
        Frag b = load_bfrag(ftg, t, kk, lane);
        tt = wmma16(aC[kk], b, tt);
      }
#pragma unroll
      for (int e = 0; e < 8; ++e)
        tg[(e + rbase) * TSTR + t * 16 + n] = tt[e];
      sched_fence();
    }
    wave_lds_fence();

    // ---- log_softmax per node row (lanes L and L+16 duplicate row L&15) ----
    {
      const float4* r4 = reinterpret_cast<const float4*>(tg + (lane & 15) * TSTR);
      float mx = -3.0e38f;
#pragma unroll
      for (int q = 0; q < 16; ++q) {
        float4 v = r4[q];
        mx = fmaxf(mx, fmaxf(fmaxf(v.x, v.y), fmaxf(v.z, v.w)));
      }
      float s = 0.f;
#pragma unroll
      for (int q = 0; q < 16; ++q) {
        float4 v = r4[q];
        s += __expf(v.x - mx) + __expf(v.y - mx) + __expf(v.z - mx) + __expf(v.w - mx);
      }
      float lse = mx + __logf(s);
      if (!hi) s_lse[wv][lane] = lse;
    }
    wave_lds_fence();

    // ---- coalesced f16 write-out (16 nodes x 64, skip padded rows) ----
#pragma unroll
    for (int it = 0; it < 16; ++it) {
      int d = it * 32 + lane;
      int mr = d >> 5, c2i = (d & 31) << 1;
      int nd = node0 + mr;
      if (nd <= last) {
        float lv = s_lse[wv][mr];
        union { f16 h[2]; unsigned int u; } pk;
        pk.h[0] = (f16)(tg[mr * TSTR + c2i    ] - lv);
        pk.h[1] = (f16)(tg[mr * TSTR + c2i + 1] - lv);
        *reinterpret_cast<unsigned int*>(out16 + (size_t)nd * 64 + c2i) = pk.u;
      }
    }
    if (isRoot && tile == 0) {
      float lv = s_lse[wv][0];
#pragma unroll
      for (int jj = 0; jj < 2; ++jj) {
        int j = jj * 32 + lane;
        rootOut[j] = tg[j] - lv;
      }
    }
    sched_fence();
  }
}

// ---------------------------------------------------------------------------
extern "C" void kernel_launch(void* const* d_in, const int* in_sizes, int n_in,
                              void* d_out, int out_size, void* d_ws, size_t ws_size,
                              hipStream_t stream) {
  (void)in_sizes; (void)n_in; (void)out_size; (void)ws_size;
  const int*   ids   = (const int*)  d_in[0];
  const float* emb   = (const float*)d_in[1];
  const float* Wih   = (const float*)d_in[2];
  const float* Whh   = (const float*)d_in[3];
  const float* bih   = (const float*)d_in[4];
  const float* bhh   = (const float*)d_in[5];
  const float* Wcomb = (const float*)d_in[6];
  const float* bcomb = (const float*)d_in[7];
  const float* Wtag  = (const float*)d_in[8];
  const float* btag  = (const float*)d_in[9];
  const float* linit = (const float*)d_in[10];
  float* out = (float*)d_out;

  char* p = (char*)d_ws;
  auto take = [&](size_t bytes) -> char* {
    char* r = p; p += (bytes + 255) & ~(size_t)255; return r;
  };
  f16*   out16  = (f16*)  take((size_t)262144 * 64 * sizeof(f16));  // internal levels 0..17
  f16*   fih    = (f16*)  take(16384 * sizeof(f16));
  f16*   fhh    = (f16*)  take(16384 * sizeof(f16));
  f16*   fch    = (f16*)  take(4096 * sizeof(f16));
  f16*   ftg    = (f16*)  take(4096 * sizeof(f16));
  float* bias1  = (float*)take(256 * sizeof(float));
  float* bias2  = (float*)take(256 * sizeof(float));
  float* c1     = (float*)take(64 * sizeof(float));
  float* cemb   = (float*)take(1664 * sizeof(float));
  f16*   leaf16 = (f16*)  take(1664 * sizeof(f16));

  prep_pack<<<160, 256, 0, stream>>>(Wih, Whh, Wcomb, Wtag, fih, fhh, fch, ftg);
  prep_small<<<1, 256, 0, stream>>>(emb, Wih, Whh, bih, bhh, Wcomb, bcomb, Wtag, btag,
                                    linit, bias1, bias2, c1, cemb, leaf16);

  for (int l = 17; l >= 0; --l) {
    int Bn = 1 << l;
    int st = Bn - 1;
    int nt = (Bn + 15) >> 4;
    int blocks = (nt + WAVES - 1) / WAVES;
    if (blocks > 1024) blocks = 1024;
    tree_level_kernel<<<blocks, 128, 0, stream>>>(
        out16, leaf16, ids, fih, fhh, fch, ftg,
        bias1, bias2, c1, cemb, btag, out,
        st, Bn, nt, (l == 17) ? 1 : 0, (l == 0) ? 1 : 0);
  }
}
